// PairScorer_7997229105355
// MI455X (gfx1250) — matrix operand order repaired and checked by reading, not compiled, not run on hardware
//
#include <hip/hip_runtime.h>
#include <hip/hip_bf16.h>

typedef __bf16 bf16;
typedef __attribute__((ext_vector_type(16))) __bf16 v16bf;
typedef __attribute__((ext_vector_type(8)))  __bf16 bf16x8;
typedef __attribute__((ext_vector_type(8)))  float  v8f;

#define NNODES 256
#define DIM    768
#define NREL   7
#define HID    150
#define HIDP   160                  // HID padded to 32 (K) / 16 (N)
#define KSTACK (8 * DIM)            // 7 relation slots + 1 root slot = 6144
#define NKK    (KSTACK / 32)        // 192 K-steps
#define NNT    (DIM / 16)           // 48 N-tiles
#define NPAIR  (NNODES * (NNODES - 1))

__device__ __forceinline__ bf16 f2bf(float f) {
  unsigned u = __builtin_bit_cast(unsigned, f);
  u += 0x7fffu + ((u >> 16) & 1u);               // round-to-nearest-even
  unsigned short s = (unsigned short)(u >> 16);
  return __builtin_bit_cast(bf16, s);
}
__device__ __forceinline__ float bf2f(bf16 b) {
  unsigned short s = __builtin_bit_cast(unsigned short, b);
  unsigned u = ((unsigned)s) << 16;
  return __builtin_bit_cast(float, u);
}

union AFrag { v16bf v; struct { bf16x8 lo, hi; } p; };

// ---------------------------------------------------------------------------
// Pack [R*D + D, D] fp32 weight stack into bf16 WMMA B-fragment layout:
// dst[((kk*NNT + nt)*32 + lane)*16 + t]  <-  W[kk*32 + ((lane&16)>>1) + (t&7) + ((t&8)<<1)][nt*16 + (lane&15)]
// ---------------------------------------------------------------------------
__global__ void pack_wstack(const float* __restrict__ w, const float* __restrict__ root,
                            bf16* __restrict__ dst) {
  int idx = blockIdx.x * blockDim.x + threadIdx.x;
  if (idx >= NKK * NNT * 512) return;
  int t    = idx & 15;
  int lane = (idx >> 4) & 31;
  int tile = idx >> 9;
  int kk = tile / NNT, nt = tile % NNT;
  int K = kk * 32 + ((lane & 16) >> 1) + (t & 7) + ((t & 8) << 1);
  int N = nt * 16 + (lane & 15);
  float v = (K < NREL * DIM) ? w[(size_t)K * DIM + N]
                             : root[(size_t)(K - NREL * DIM) * DIM + N];
  dst[idx] = f2bf(v);
}

// Pack classifier W2 [150,150] fp32 -> padded [160,160] bf16, B-fragment layout.
__global__ void pack_w2(const float* __restrict__ w2, bf16* __restrict__ dst) {
  const int NT2 = HIDP / 16;                     // 10
  int idx = blockIdx.x * blockDim.x + threadIdx.x;
  if (idx >= (HIDP / 32) * NT2 * 512) return;
  int t = idx & 15, lane = (idx >> 4) & 31, tile = idx >> 9;
  int kk = tile / NT2, nt = tile % NT2;
  int K = kk * 32 + ((lane & 16) >> 1) + (t & 7) + ((t & 8) << 1);
  int N = nt * 16 + (lane & 15);
  float v = (K < HID && N < HID) ? w2[K * HID + N] : 0.f;
  dst[idx] = f2bf(v);
}

// ---------------------------------------------------------------------------
// RGCN mean-aggregation. One block per destination node k. Produces the
// stacked GEMM input row: [7 relation means | x_k] as bf16 [KSTACK].
// keep = (label != 6); etype contributes nothing when label==6.
// ---------------------------------------------------------------------------
__global__ void rgcn_aggregate(const float* __restrict__ x, const int* __restrict__ labels,
                               bf16* __restrict__ aggA) {
  __shared__ float acc[NREL * DIM];
  __shared__ float cnt[NREL];
  int k = blockIdx.x, tid = threadIdx.x;
  for (int d = tid; d < NREL * DIM; d += blockDim.x) acc[d] = 0.f;
  if (tid < NREL) cnt[tid] = 0.f;
  __syncthreads();
  for (int i = 0; i < NNODES; ++i) {
    if (i == k) continue;
    int p = i * (NNODES - 1) + k - (k > i ? 1 : 0);  // ordered-pair index (i -> k)
    int l = labels[p];                               // uniform -> scalar load
    if (l == NREL - 1) continue;                     // NONE edges dropped
    const float* xi = x + (size_t)i * DIM;
    for (int d = tid; d < DIM; d += blockDim.x) acc[l * DIM + d] += xi[d];
    if (tid == 0) cnt[l] += 1.f;
  }
  __syncthreads();
  bf16* row = aggA + (size_t)k * KSTACK;
  for (int d = tid; d < DIM; d += blockDim.x) {
    for (int r = 0; r < NREL; ++r) {
      float c = cnt[r]; c = c < 1.f ? 1.f : c;
      row[r * DIM + d] = f2bf(acc[r * DIM + d] / c);
    }
    row[NREL * DIM + d] = f2bf(x[(size_t)k * DIM + d]);   // root slot
  }
}

// ---------------------------------------------------------------------------
// out[256,768] = A[256,6144]_bf16 @ Wstack[6144,768]_bf16 + bias (+relu)
// grid (16 M-tiles, 6); 8 waves/block, one N-tile per wave, WMMA over 192 K-steps.
// ---------------------------------------------------------------------------
__global__ void rgcn_gemm(const bf16* __restrict__ A, const bf16* __restrict__ Bp,
                          const float* __restrict__ bias, float* __restrict__ out,
                          int relu) {
  int lane = threadIdx.x & 31;
  int wave = threadIdx.x >> 5;
  int mt = blockIdx.x;
  int nt = blockIdx.y * 8 + wave;                 // 0..47
  int mrow  = mt * 16 + (lane & 15);
  int abase = ((lane >> 4) & 1) * 8;              // A-frag K base per half-wave
  v8f acc = {};
  const bf16* arow = A + (size_t)mrow * KSTACK;
  for (int kk = 0; kk < NKK; ++kk) {
    AFrag a, b;
    a.p.lo = *(const bf16x8*)(arow + kk * 32 + abase);
    a.p.hi = *(const bf16x8*)(arow + kk * 32 + 16 + abase);
    const bf16* bptr = Bp + (((size_t)kk * NNT + nt) * 32 + lane) * 16;
    b.p.lo = *(const bf16x8*)(bptr);
    b.p.hi = *(const bf16x8*)(bptr + 8);
    acc = __builtin_amdgcn_wmma_f32_16x16x32_bf16(false, a.v, false, b.v,
                                                  (short)0, acc, false, false);
  }
  int col = nt * 16 + (lane & 15);
  float bv = bias[col];
  int rbase = mt * 16 + ((lane >> 4) & 1) * 8;    // C layout: M = v + 8*(lane>=16)
#pragma unroll
  for (int v = 0; v < 8; ++v) {
    float val = acc[v] + bv;
    if (relu) val = val > 0.f ? val : 0.f;
    out[(size_t)(rbase + v) * DIM + col] = val;
  }
}

// Node-level factorization of the pair concat GEMM:
// An[n,h] = out[n,:] @ W1[0:768, h];  Bn[n,h] = out[n,:] @ W1[768:1536, h]
__global__ void node_proj(const float* __restrict__ x, const float* __restrict__ w1,
                          float* __restrict__ An, float* __restrict__ Bn) {
  int idx = blockIdx.x * blockDim.x + threadIdx.x;
  if (idx >= 2 * NNODES * HID) return;
  int which = idx / (NNODES * HID);
  int rem = idx % (NNODES * HID);
  int n = rem / HID, h = rem % HID;
  const float* xr = x + (size_t)n * DIM;
  const float* wc = w1 + (which ? DIM * HID : 0) + h;
  float s = 0.f;
  for (int d = 0; d < DIM; ++d) s += xr[d] * wc[(size_t)d * HID];
  (which ? Bn : An)[n * HID + h] = s;
}

// ---------------------------------------------------------------------------
// Per-pair MLP. 128 pairs/block; each wave owns a 16-pair M-tile.
// h1 built directly in A-fragment registers; h2 = relu(h1@W2+b2) via WMMA;
// logits = h2@W3+b3 via VALU from LDS.
// ---------------------------------------------------------------------------
__global__ void pair_mlp(const float* __restrict__ An, const float* __restrict__ Bn,
                         const float* __restrict__ b1, const bf16* __restrict__ w2p,
                         const float* __restrict__ b2, const float* __restrict__ w3,
                         const float* __restrict__ b3, float* __restrict__ out) {
  __shared__ bf16 h2s[128 * HIDP];                // 40 KB
  int lane = threadIdx.x & 31;
  int wave = threadIdx.x >> 5;
  int prow = wave * 16 + (lane & 15);             // this lane's pair row (M)
  int p = blockIdx.x * 128 + prow;
  int i = p / (NNODES - 1);
  int q = p % (NNODES - 1);
  int k = q + (q >= i ? 1 : 0);
  const float* Ar = An + i * HID;
  const float* Br = Bn + k * HID;
  int abase = ((lane >> 4) & 1) * 8;
  v8f acc[10] = {};
  for (int kk = 0; kk < 5; ++kk) {
    AFrag a;
#pragma unroll
    for (int t = 0; t < 8; ++t) {
      int h0 = kk * 32 + abase + t;
      int h1 = kk * 32 + 16 + abase + t;
      float v0 = 0.f, v1 = 0.f;
      if (h0 < HID) { v0 = Ar[h0] + Br[h0] + b1[h0]; v0 = v0 > 0.f ? v0 : 0.f; }
      if (h1 < HID) { v1 = Ar[h1] + Br[h1] + b1[h1]; v1 = v1 > 0.f ? v1 : 0.f; }
      a.p.lo[t] = f2bf(v0);
      a.p.hi[t] = f2bf(v1);
    }
#pragma unroll
    for (int nt = 0; nt < 10; ++nt) {
      AFrag b;
      const bf16* bptr = w2p + (((size_t)kk * 10 + nt) * 32 + lane) * 16;
      b.p.lo = *(const bf16x8*)bptr;
      b.p.hi = *(const bf16x8*)(bptr + 8);
      acc[nt] = __builtin_amdgcn_wmma_f32_16x16x32_bf16(false, a.v, false, b.v,
                                                        (short)0, acc[nt], false, false);
    }
  }
  int rbase = wave * 16 + ((lane >> 4) & 1) * 8;
#pragma unroll
  for (int nt = 0; nt < 10; ++nt) {
    int col = nt * 16 + (lane & 15);
    float bv = (col < HID) ? b2[col] : 0.f;
#pragma unroll
    for (int v = 0; v < 8; ++v) {
      float val = acc[nt][v] + bv;
      val = val > 0.f ? val : 0.f;
      h2s[(rbase + v) * HIDP + col] = f2bf(val);
    }
  }
  __syncthreads();
  for (int e = threadIdx.x; e < 128 * NREL; e += blockDim.x) {
    int row = e / NREL, o = e % NREL;
    float s = b3[o];
    const bf16* hr = h2s + row * HIDP;
    for (int h = 0; h < HID; ++h) s += bf2f(hr[h]) * w3[h * NREL + o];
    out[((size_t)blockIdx.x * 128 + row) * NREL + o] = s;
  }
}

extern "C" void kernel_launch(void* const* d_in, const int* in_sizes, int n_in,
                              void* d_out, int out_size, void* d_ws, size_t ws_size,
                              hipStream_t stream) {
  (void)in_sizes; (void)n_in; (void)out_size; (void)ws_size;
  const float* embed  = (const float*)d_in[0];
  const int*   labels = (const int*)  d_in[1];
  const float* c1w = (const float*)d_in[2];
  const float* c1r = (const float*)d_in[3];
  const float* c1b = (const float*)d_in[4];
  const float* c2w = (const float*)d_in[5];
  const float* c2r = (const float*)d_in[6];
  const float* c2b = (const float*)d_in[7];
  // d_in[8..13]: baseline scorer weights — its output is discarded by the
  // reference, so it is dead code and skipped entirely.
  const float* w1 = (const float*)d_in[14];
  const float* b1 = (const float*)d_in[15];
  const float* w2 = (const float*)d_in[16];
  const float* b2 = (const float*)d_in[17];
  const float* w3 = (const float*)d_in[18];
  const float* b3 = (const float*)d_in[19];
  float* outp = (float*)d_out;

  char* ws = (char*)d_ws;
  size_t off = 0;
  auto alloc = [&](size_t bytes) -> void* {
    void* q = ws + off;
    off = (off + bytes + 255) & ~(size_t)255;
    return q;
  };
  bf16*  wpk1 = (bf16*) alloc((size_t)KSTACK * DIM * 2);   // 9.4 MB
  bf16*  wpk2 = (bf16*) alloc((size_t)KSTACK * DIM * 2);   // 9.4 MB
  bf16*  w2pk = (bf16*) alloc((size_t)HIDP * HIDP * 2);    // 51 KB
  bf16*  aggA = (bf16*) alloc((size_t)NNODES * KSTACK * 2);// 3.1 MB
  float* x1   = (float*)alloc((size_t)NNODES * DIM * 4);
  float* x2   = (float*)alloc((size_t)NNODES * DIM * 4);
  float* Anod = (float*)alloc((size_t)NNODES * HID * 4);
  float* Bnod = (float*)alloc((size_t)NNODES * HID * 4);

  int packE = NKK * NNT * 512;
  pack_wstack<<<(packE + 255) / 256, 256, 0, stream>>>(c1w, c1r, wpk1);
  pack_wstack<<<(packE + 255) / 256, 256, 0, stream>>>(c2w, c2r, wpk2);
  int p2e = (HIDP / 32) * (HIDP / 16) * 512;
  pack_w2<<<(p2e + 255) / 256, 256, 0, stream>>>(w2, w2pk);

  rgcn_aggregate<<<NNODES, 256, 0, stream>>>(embed, labels, aggA);
  rgcn_gemm<<<dim3(16, 6), 256, 0, stream>>>(aggA, wpk1, c1b, x1, 1);
  rgcn_aggregate<<<NNODES, 256, 0, stream>>>(x1, labels, aggA);
  rgcn_gemm<<<dim3(16, 6), 256, 0, stream>>>(aggA, wpk2, c2b, x2, 0);

  node_proj<<<(2 * NNODES * HID + 255) / 256, 256, 0, stream>>>(x2, w1, Anod, Bnod);
  pair_mlp<<<NPAIR / 128, 256, 0, stream>>>(Anod, Bnod, b1, w2pk, b2, w3, b3, outp);
}
